// Graphormer_4071628996729
// MI455X (gfx1250) — matrix operand order repaired and checked by reading, not compiled, not run on hardware
//
#include <hip/hip_runtime.h>
#include <hip/hip_bf16.h>
#include <stdint.h>

// ---------------------------------------------------------------------------
// Types for CDNA5 WMMA (wave32): v_wmma_f32_16x16x32_bf16
// ---------------------------------------------------------------------------
typedef __attribute__((ext_vector_type(16))) __bf16 v16bf;
typedef __attribute__((ext_vector_type(8)))  float  v8f;

union BF16Frag { v16bf v; uint4 u[2]; };

// ---------------------------------------------------------------------------
// Small utility kernels
// ---------------------------------------------------------------------------
__global__ void fill_u32_kernel(uint32_t* __restrict__ p, uint32_t val, int n) {
  int i = blockIdx.x * blockDim.x + threadIdx.x;
  if (i < n) p[i] = val;
}

__global__ void f32_to_bf16_kernel(const float* __restrict__ in,
                                   __hip_bfloat16* __restrict__ out, int n) {
  int i = blockIdx.x * blockDim.x + threadIdx.x;
  if (i < n) out[i] = __float2bfloat16(in[i]);
}

// W [K, N] fp32 row-major  ->  Wt [N, K] bf16 row-major (K contiguous)
__global__ void transpose_f32_to_bf16_kernel(const float* __restrict__ W,
                                             __hip_bfloat16* __restrict__ Wt,
                                             int K, int N) {
  int i = blockIdx.x * blockDim.x + threadIdx.x;
  if (i < K * N) {
    int k = i / N;
    int n = i - k * N;
    Wt[(size_t)n * K + k] = __float2bfloat16(W[i]);
  }
}

// ---------------------------------------------------------------------------
// WMMA GEMM:  C[M,Nout] = A[M,K](bf16) * Bt[Nout,K](bf16)^T + bias[Nout]
// Block = 64 threads (2 waves). Each wave computes a 32x64 tile:
//   2 A fragments (rows mBase..+15, mBase+16..+31) x 4 B fragments
//   -> 8 v_wmma_f32_16x16x32_bf16 per K-step of 32.
// Each B fragment feeds two WMMAs (2x the math per load vs a 16x64 tile).
// Fragment layouts per CDNA5 ISA 7.12.2 (16-bit A 16x32; B column-per-lane).
// ---------------------------------------------------------------------------
template <int K>
__global__ void wmma_gemm_bias_kernel(const __hip_bfloat16* __restrict__ A,
                                      const __hip_bfloat16* __restrict__ Bt,
                                      const float* __restrict__ bias,
                                      float* __restrict__ C,
                                      int M, int Nout) {
  const int lane  = threadIdx.x & 31;
  const int wave  = threadIdx.x >> 5;
  const int mBase = blockIdx.x << 5;                 // 32 rows per block
  const int nBase = ((blockIdx.y << 1) + wave) << 6; // 64 cols per wave
  const int mrow  = lane & 15;
  const int h     = lane >> 4;                       // lane half (0/1)

  int ar0 = mBase + mrow;        if (ar0 >= M) ar0 = M - 1;  // clamp tail rows
  int ar1 = mBase + 16 + mrow;   if (ar1 >= M) ar1 = M - 1;
  const __hip_bfloat16* arow0 = A + (size_t)ar0 * K;
  const __hip_bfloat16* arow1 = A + (size_t)ar1 * K;
  const __hip_bfloat16* bcol0 = Bt + (size_t)(nBase + (lane & 15)) * K;

  v8f acc[2][4] = {};

#pragma unroll
  for (int k0 = 0; k0 < K; k0 += 32) {
    // A fragments: lane half h holds K = k0+8h..+7 and k0+16+8h..+7
    BF16Frag a0, a1;
    a0.u[0] = *(const uint4*)(arow0 + k0 + 8 * h);
    a0.u[1] = *(const uint4*)(arow0 + k0 + 16 + 8 * h);
    a1.u[0] = *(const uint4*)(arow1 + k0 + 8 * h);
    a1.u[1] = *(const uint4*)(arow1 + k0 + 16 + 8 * h);
#pragma unroll
    for (int t = 0; t < 4; ++t) {
      // B fragment t: lane = column (nBase + t*16 + lane%16), K = k0+8h..+7 and k0+16+8h..+7
      const __hip_bfloat16* bp = bcol0 + (size_t)(t * 16) * K + k0 + 8 * h;
      BF16Frag bf;
      bf.u[0] = *(const uint4*)(bp);
      bf.u[1] = *(const uint4*)(bp + 16);
      acc[0][t] = __builtin_amdgcn_wmma_f32_16x16x32_bf16(
          false, a0.v, false, bf.v, (short)0, acc[0][t], false, false);
      acc[1][t] = __builtin_amdgcn_wmma_f32_16x16x32_bf16(
          false, a1.v, false, bf.v, (short)0, acc[1][t], false, false);
    }
  }

  // D layout: VGPR r -> row (tileBase + r + 8h) ; lane%16 -> column
#pragma unroll
  for (int t = 0; t < 4; ++t) {
    const int n  = nBase + t * 16 + (lane & 15);
    const float b = bias[n];
#pragma unroll
    for (int g = 0; g < 2; ++g) {
      const int mb = mBase + 16 * g;
#pragma unroll
      for (int r = 0; r < 8; ++r) {
        const int m = mb + r + 8 * h;
        if (m < M) C[(size_t)m * Nout + n] = acc[g][t][r] + b;
      }
    }
  }
}

// ---------------------------------------------------------------------------
// Edge kernels (C = 128 per head, fixed).  One wave per (edge, head).
// ---------------------------------------------------------------------------
__device__ __forceinline__ void atomicMaxF32(float* addr, float val) {
  if (val >= 0.0f) atomicMax((int*)addr, __float_as_int(val));
  else             atomicMin((unsigned int*)addr, __float_as_uint(val));
}

template <int H>
__global__ void edge_alpha_kernel(const float* __restrict__ q,   // [N,H,128]
                                  const float* __restrict__ k,   // [N,H,128]
                                  const int* __restrict__ src,
                                  const int* __restrict__ dst,
                                  float* __restrict__ alpha,     // [E,H]
                                  float* __restrict__ amax,      // [N,H]
                                  int E, float scale) {
  const int wid  = (blockIdx.x * blockDim.x + threadIdx.x) >> 5;
  const int lane = threadIdx.x & 31;
  if (wid >= E * H) return;
  const int e = wid / H, h = wid - e * H;
  const int s = src[e], d = dst[e];
  const float4 qa = *(const float4*)(q + ((size_t)d * H + h) * 128 + lane * 4);
  const float4 kb = *(const float4*)(k + ((size_t)s * H + h) * 128 + lane * 4);
  float part = qa.x * kb.x + qa.y * kb.y + qa.z * kb.z + qa.w * kb.w;
#pragma unroll
  for (int off = 16; off > 0; off >>= 1) part += __shfl_xor(part, off, 32);
  if (lane == 0) {
    const float a = part * scale;
    alpha[(size_t)e * H + h] = a;
    atomicMaxF32(&amax[(size_t)d * H + h], a);
  }
}

template <int H>
__global__ void edge_softmax_num_kernel(const int* __restrict__ dst,
                                        float* __restrict__ alpha,   // in: logits, out: p
                                        const float* __restrict__ amax,
                                        float* __restrict__ denom,
                                        int E) {
  const int i = blockIdx.x * blockDim.x + threadIdx.x;
  if (i >= E * H) return;
  const int e = i / H, h = i - e * H;
  const int d = dst[e];
  const float p = __expf(alpha[i] - amax[(size_t)d * H + h]);
  alpha[i] = p;
  atomicAdd(&denom[(size_t)d * H + h], p);
}

template <int H>
__global__ void edge_scatter_kernel(const int* __restrict__ src,
                                    const int* __restrict__ dst,
                                    const float* __restrict__ alpha, // p [E,H]
                                    const float* __restrict__ v,     // [N,H,128]
                                    float* __restrict__ attn,        // [N,H,128]
                                    int E) {
  const int wid  = (blockIdx.x * blockDim.x + threadIdx.x) >> 5;
  const int lane = threadIdx.x & 31;
  if (wid >= E * H) return;
  const int e = wid / H, h = wid - e * H;
  const int s = src[e], d = dst[e];
  const float p = alpha[(size_t)e * H + h];
  const float4 vv = *(const float4*)(v + ((size_t)s * H + h) * 128 + lane * 4);
  float* o = attn + ((size_t)d * H + h) * 128 + lane * 4;
  atomicAdd(o + 0, p * vv.x);
  atomicAdd(o + 1, p * vv.y);
  atomicAdd(o + 2, p * vv.z);
  atomicAdd(o + 3, p * vv.w);
}

// layer1 finalize: h1 = relu(attn/denom + skip), concat heads (H1=4, 512 wide)
__global__ void finalize1_kernel(const float* __restrict__ attn,
                                 const float* __restrict__ denom, // [N,4]
                                 const float* __restrict__ skip,  // [N,512]
                                 float* __restrict__ out,         // [N,512]
                                 int total) {
  const int i = blockIdx.x * blockDim.x + threadIdx.x;
  if (i >= total) return;
  const int n = i >> 9;            // /512
  const int h = (i & 511) >> 7;    // head
  const float d = denom[(size_t)n * 4 + h] + 1e-16f;
  out[i] = fmaxf(attn[i] / d + skip[i], 0.0f);
}

// layer2 finalize: out = mean over 2 heads of attn/denom + skip (128 wide)
__global__ void finalize2_kernel(const float* __restrict__ attn,  // [N,2,128]
                                 const float* __restrict__ denom, // [N,2]
                                 const float* __restrict__ skip,  // [N,128]
                                 float* __restrict__ out,         // [N,128]
                                 int total) {
  const int i = blockIdx.x * blockDim.x + threadIdx.x;
  if (i >= total) return;
  const int n = i >> 7;
  const int c = i & 127;
  const float a0 = attn[(size_t)n * 256 + c]       / (denom[(size_t)n * 2 + 0] + 1e-16f);
  const float a1 = attn[(size_t)n * 256 + 128 + c] / (denom[(size_t)n * 2 + 1] + 1e-16f);
  out[i] = 0.5f * (a0 + a1) + skip[i];
}

// ---------------------------------------------------------------------------
// Host orchestration
// ---------------------------------------------------------------------------
extern "C" void kernel_launch(void* const* d_in, const int* in_sizes, int n_in,
                              void* d_out, int out_size, void* d_ws, size_t ws_size,
                              hipStream_t stream) {
  const int IN  = 128;
  const int HC1 = 512;   // H1*C1
  const int H1  = 4;
  const int HC2 = 256;   // H2*C2
  const int H2  = 2;
  const int C2o = 128;   // Ws2 output width

  const int N = in_sizes[0] / IN;
  const int E = in_sizes[1] / 2;

  const float* x   = (const float*)d_in[0];
  const int*   ei  = (const int*)d_in[1];
  const int*   src = ei;
  const int*   dst = ei + E;
  const float* Wq1 = (const float*)d_in[2];  const float* bq1 = (const float*)d_in[3];
  const float* Wk1 = (const float*)d_in[4];  const float* bk1 = (const float*)d_in[5];
  const float* Wv1 = (const float*)d_in[6];  const float* bv1 = (const float*)d_in[7];
  const float* Ws1 = (const float*)d_in[8];  const float* bs1 = (const float*)d_in[9];
  const float* Wq2 = (const float*)d_in[10]; const float* bq2 = (const float*)d_in[11];
  const float* Wk2 = (const float*)d_in[12]; const float* bk2 = (const float*)d_in[13];
  const float* Wv2 = (const float*)d_in[14]; const float* bv2 = (const float*)d_in[15];
  const float* Ws2 = (const float*)d_in[16]; const float* bs2 = (const float*)d_in[17];
  float* out = (float*)d_out;

  // Workspace carving (256B aligned)
  char* base = (char*)d_ws;
  size_t cur = 0;
  auto alloc = [&](size_t bytes) -> char* {
    cur = (cur + 255) & ~(size_t)255;
    char* p = base + cur;
    cur += bytes;
    return p;
  };

  __hip_bfloat16* xb   = (__hip_bfloat16*)alloc((size_t)N * IN * 2);
  __hip_bfloat16* wq1t = (__hip_bfloat16*)alloc((size_t)IN * HC1 * 2);
  __hip_bfloat16* wk1t = (__hip_bfloat16*)alloc((size_t)IN * HC1 * 2);
  __hip_bfloat16* wv1t = (__hip_bfloat16*)alloc((size_t)IN * HC1 * 2);
  __hip_bfloat16* ws1t = (__hip_bfloat16*)alloc((size_t)IN * HC1 * 2);
  float* q1    = (float*)alloc((size_t)N * HC1 * 4);
  float* k1    = (float*)alloc((size_t)N * HC1 * 4);
  float* v1    = (float*)alloc((size_t)N * HC1 * 4);
  float* s1    = (float*)alloc((size_t)N * HC1 * 4);
  float* al1   = (float*)alloc((size_t)E * H1 * 4);
  float* amax1 = (float*)alloc((size_t)N * H1 * 4);
  float* den1  = (float*)alloc((size_t)N * H1 * 4);
  float* attn1 = (float*)alloc((size_t)N * HC1 * 4);
  __hip_bfloat16* h1b  = (__hip_bfloat16*)alloc((size_t)N * HC1 * 2);
  __hip_bfloat16* wq2t = (__hip_bfloat16*)alloc((size_t)HC1 * HC2 * 2);
  __hip_bfloat16* wk2t = (__hip_bfloat16*)alloc((size_t)HC1 * HC2 * 2);
  __hip_bfloat16* wv2t = (__hip_bfloat16*)alloc((size_t)HC1 * HC2 * 2);
  __hip_bfloat16* ws2t = (__hip_bfloat16*)alloc((size_t)HC1 * C2o * 2);
  float* q2    = (float*)alloc((size_t)N * HC2 * 4);
  float* k2    = (float*)alloc((size_t)N * HC2 * 4);
  float* v2    = (float*)alloc((size_t)N * HC2 * 4);
  float* s2    = (float*)alloc((size_t)N * C2o * 4);
  float* al2   = (float*)alloc((size_t)E * H2 * 4);
  float* amax2 = (float*)alloc((size_t)N * H2 * 4);
  float* den2  = (float*)alloc((size_t)N * H2 * 4);
  float* attn2 = (float*)alloc((size_t)N * HC2 * 4);
  (void)ws_size; (void)n_in; (void)out_size;

  const int T = 256;
  auto blk = [](int n, int t) { return (n + t - 1) / t; };
  const int M32 = (N + 31) / 32;              // 32-row panels per GEMM block
  const float scale = 0.08838834764831845f;   // 1/sqrt(128)

  // ---- layer 1 ----
  f32_to_bf16_kernel<<<blk(N * IN, T), T, 0, stream>>>(x, xb, N * IN);
  transpose_f32_to_bf16_kernel<<<blk(IN * HC1, T), T, 0, stream>>>(Wq1, wq1t, IN, HC1);
  transpose_f32_to_bf16_kernel<<<blk(IN * HC1, T), T, 0, stream>>>(Wk1, wk1t, IN, HC1);
  transpose_f32_to_bf16_kernel<<<blk(IN * HC1, T), T, 0, stream>>>(Wv1, wv1t, IN, HC1);
  transpose_f32_to_bf16_kernel<<<blk(IN * HC1, T), T, 0, stream>>>(Ws1, ws1t, IN, HC1);

  dim3 g1(M32, HC1 / 128);
  wmma_gemm_bias_kernel<128><<<g1, 64, 0, stream>>>(xb, wq1t, bq1, q1, N, HC1);
  wmma_gemm_bias_kernel<128><<<g1, 64, 0, stream>>>(xb, wk1t, bk1, k1, N, HC1);
  wmma_gemm_bias_kernel<128><<<g1, 64, 0, stream>>>(xb, wv1t, bv1, v1, N, HC1);
  wmma_gemm_bias_kernel<128><<<g1, 64, 0, stream>>>(xb, ws1t, bs1, s1, N, HC1);

  fill_u32_kernel<<<blk(N * H1, T), T, 0, stream>>>((uint32_t*)amax1, 0xFF800000u, N * H1);
  fill_u32_kernel<<<blk(N * H1, T), T, 0, stream>>>((uint32_t*)den1, 0u, N * H1);
  fill_u32_kernel<<<blk(N * HC1, T), T, 0, stream>>>((uint32_t*)attn1, 0u, N * HC1);

  edge_alpha_kernel<4><<<blk(E * H1 * 32, T), T, 0, stream>>>(q1, k1, src, dst, al1, amax1, E, scale);
  edge_softmax_num_kernel<4><<<blk(E * H1, T), T, 0, stream>>>(dst, al1, amax1, den1, E);
  edge_scatter_kernel<4><<<blk(E * H1 * 32, T), T, 0, stream>>>(src, dst, al1, v1, attn1, E);
  finalize1_kernel<<<blk(N * HC1, T), T, 0, stream>>>(attn1, den1, s1, attn1, N * HC1);
  f32_to_bf16_kernel<<<blk(N * HC1, T), T, 0, stream>>>(attn1, h1b, N * HC1);

  // ---- layer 2 ----
  transpose_f32_to_bf16_kernel<<<blk(HC1 * HC2, T), T, 0, stream>>>(Wq2, wq2t, HC1, HC2);
  transpose_f32_to_bf16_kernel<<<blk(HC1 * HC2, T), T, 0, stream>>>(Wk2, wk2t, HC1, HC2);
  transpose_f32_to_bf16_kernel<<<blk(HC1 * HC2, T), T, 0, stream>>>(Wv2, wv2t, HC1, HC2);
  transpose_f32_to_bf16_kernel<<<blk(HC1 * C2o, T), T, 0, stream>>>(Ws2, ws2t, HC1, C2o);

  dim3 g2(M32, HC2 / 128);
  wmma_gemm_bias_kernel<512><<<g2, 64, 0, stream>>>(h1b, wq2t, bq2, q2, N, HC2);
  wmma_gemm_bias_kernel<512><<<g2, 64, 0, stream>>>(h1b, wk2t, bk2, k2, N, HC2);
  wmma_gemm_bias_kernel<512><<<g2, 64, 0, stream>>>(h1b, wv2t, bv2, v2, N, HC2);
  dim3 g2s(M32, C2o / 128);
  wmma_gemm_bias_kernel<512><<<g2s, 64, 0, stream>>>(h1b, ws2t, bs2, s2, N, C2o);

  fill_u32_kernel<<<blk(N * H2, T), T, 0, stream>>>((uint32_t*)amax2, 0xFF800000u, N * H2);
  fill_u32_kernel<<<blk(N * H2, T), T, 0, stream>>>((uint32_t*)den2, 0u, N * H2);
  fill_u32_kernel<<<blk(N * HC2, T), T, 0, stream>>>((uint32_t*)attn2, 0u, N * HC2);

  edge_alpha_kernel<2><<<blk(E * H2 * 32, T), T, 0, stream>>>(q2, k2, src, dst, al2, amax2, E, scale);
  edge_softmax_num_kernel<2><<<blk(E * H2, T), T, 0, stream>>>(dst, al2, amax2, den2, E);
  edge_scatter_kernel<2><<<blk(E * H2 * 32, T), T, 0, stream>>>(src, dst, al2, v2, attn2, E);
  finalize2_kernel<<<blk(N * C2o, T), T, 0, stream>>>(attn2, den2, s2, out, N * C2o);
}
